// MSTFormer_53403623358556
// MI455X (gfx1250) — matrix operand order repaired and checked by reading, not compiled, not run on hardware
//
#include <hip/hip_runtime.h>
#include <hip/hip_bf16.h>
#include <cmath>

// ---------------------------------------------------------------------------
// MSTFormer (Informer-style) forward for gfx1250.
// Dense GEMMs -> v_wmma_f32_16x16x32_f16 (128x64 block tile, 32x32 register
// tile per wave, double-buffered LDS) ; everything else VALU kernels.
// ---------------------------------------------------------------------------

typedef __attribute__((ext_vector_type(16))) _Float16 v16h;
typedef __attribute__((ext_vector_type(8)))  _Float16 v8h;
typedef __attribute__((ext_vector_type(2)))  _Float16 v2h;
typedef __attribute__((ext_vector_type(8)))  float    v8f;

#define LDK 40            // padded LDS K-stride in halves (80 B rows, 16B aligned)
#define ASZ (128 * LDK)   // one A buffer (halves)
#define BSZ (64 * LDK)    // one B buffer (halves)

// one double-rate step: 2 A-frags x 2 B-frags -> 4 WMMAs
__device__ __forceinline__ void wmma_step(const _Float16* __restrict__ As,
                                          const _Float16* __restrict__ Bs,
                                          int wm, int wn, int r, int hh,
                                          v8f& a00, v8f& a01, v8f& a10, v8f& a11)
{
    // A fragments (16x32 f16): VGPR0-3 K=hh*8..+7, VGPR4-7 K=16+hh*8..+7
    v8h l0 = *reinterpret_cast<const v8h*>(&As[(wm + r) * LDK + hh * 8]);
    v8h l1 = *reinterpret_cast<const v8h*>(&As[(wm + r) * LDK + 16 + hh * 8]);
    v8h l2 = *reinterpret_cast<const v8h*>(&As[(wm + 16 + r) * LDK + hh * 8]);
    v8h l3 = *reinterpret_cast<const v8h*>(&As[(wm + 16 + r) * LDK + 16 + hh * 8]);
    v16h af0, af1;
    #pragma unroll
    for (int i = 0; i < 8; ++i) {
        af0[i] = l0[i]; af0[8 + i] = l1[i];
        af1[i] = l2[i]; af1[8 + i] = l3[i];
    }
    // B fragments (32x16 f16): VGPR0-3 K=hh*8..+7, VGPR4-7 K=16+hh*8..+7
    const int kb = hh * 8;
    v8h b0 = *reinterpret_cast<const v8h*>(&Bs[(wn + r) * LDK + kb]);
    v8h b1 = *reinterpret_cast<const v8h*>(&Bs[(wn + r) * LDK + kb + 16]);
    v8h b2 = *reinterpret_cast<const v8h*>(&Bs[(wn + 16 + r) * LDK + kb]);
    v8h b3 = *reinterpret_cast<const v8h*>(&Bs[(wn + 16 + r) * LDK + kb + 16]);
    v16h bf0, bf1;
    #pragma unroll
    for (int i = 0; i < 8; ++i) {
        bf0[i] = b0[i]; bf0[8 + i] = b1[i];
        bf1[i] = b2[i]; bf1[8 + i] = b3[i];
    }
    a00 = __builtin_amdgcn_wmma_f32_16x16x32_f16(false, af0, false, bf0, (short)0, a00, false, false);
    a01 = __builtin_amdgcn_wmma_f32_16x16x32_f16(false, af0, false, bf1, (short)0, a01, false, false);
    a10 = __builtin_amdgcn_wmma_f32_16x16x32_f16(false, af1, false, bf0, (short)0, a10, false, false);
    a11 = __builtin_amdgcn_wmma_f32_16x16x32_f16(false, af1, false, bf1, (short)0, a11, false, false);
}

// ======================= WMMA GEMM =========================================
// C[M,N] = act( A[M,K] @ W[K,N] )
// A row remap: arow = m + (m / rows_per_batch) * extra_rows (implicit im2col
// over circular-padded buffers); lda = A row stride in floats.
// M % 128 == 0, N % 64 == 0, K % 32 == 0 (guaranteed by caller).
__global__ __launch_bounds__(256)
void k_gemm(const float* __restrict__ A, const float* __restrict__ W,
            float* __restrict__ C, int M, int N, int K, int lda,
            int rows_per_batch, int extra_rows, int act)
{
    __shared__ __align__(16) _Float16 As[2 * ASZ];
    __shared__ __align__(16) _Float16 Bs[2 * BSZ];

    const int m0 = blockIdx.y * 128;
    const int n0 = blockIdx.x * 64;
    const int tid  = threadIdx.x;
    const int wave = tid >> 5;
    const int lane = tid & 31;
    const int wm = (wave & 3) * 32;     // wave M strip (two 16-row sub-tiles)
    const int wn = (wave >> 2) * 32;    // wave N strip (two 16-col sub-tiles)
    const int r  = lane & 15;
    const int hh = lane >> 4;

    v8f a00 = {}; v8f a01 = {}; v8f a10 = {}; v8f a11 = {};

    // A staging: 16 f32 per thread -> 16 f16 (one row chunk)
    const int arow = tid >> 1;              // 0..127
    const int acol = (tid & 1) * 16;        // 0 or 16
    const long am    = (long)m0 + arow;
    const long agrow = am + (am / rows_per_batch) * extra_rows;
    const float* aptr = A + agrow * (long)lda + acol;

    // B staging: thread covers column n, 4 packed k-pairs (transposed store)
    const int bn = tid & 63;                // 0..63
    const int kp = tid >> 6;                // 0..3

    auto stage = [&](int bsel, int k0) {
        // ---- A tile [128][32] f16, row-major
        const float4 q0 = *reinterpret_cast<const float4*>(aptr + k0);
        const float4 q1 = *reinterpret_cast<const float4*>(aptr + k0 + 4);
        const float4 q2 = *reinterpret_cast<const float4*>(aptr + k0 + 8);
        const float4 q3 = *reinterpret_cast<const float4*>(aptr + k0 + 12);
        v8h h0, h1;
        h0[0] = (_Float16)q0.x; h0[1] = (_Float16)q0.y;
        h0[2] = (_Float16)q0.z; h0[3] = (_Float16)q0.w;
        h0[4] = (_Float16)q1.x; h0[5] = (_Float16)q1.y;
        h0[6] = (_Float16)q1.z; h0[7] = (_Float16)q1.w;
        h1[0] = (_Float16)q2.x; h1[1] = (_Float16)q2.y;
        h1[2] = (_Float16)q2.z; h1[3] = (_Float16)q2.w;
        h1[4] = (_Float16)q3.x; h1[5] = (_Float16)q3.y;
        h1[6] = (_Float16)q3.z; h1[7] = (_Float16)q3.w;
        _Float16* as = &As[bsel * ASZ + arow * LDK + acol];
        *reinterpret_cast<v8h*>(as)     = h0;
        *reinterpret_cast<v8h*>(as + 8) = h1;
        // ---- B tile transposed: Bs[n][k], packed k-pairs as b32 stores
        #pragma unroll
        for (int i = 0; i < 4; ++i) {
            const int kk = kp * 2 + i * 8;
            const float* wp = W + (size_t)(k0 + kk) * N + n0 + bn;
            v2h p;
            p[0] = (_Float16)wp[0];
            p[1] = (_Float16)wp[N];
            *reinterpret_cast<v2h*>(&Bs[bsel * BSZ + bn * LDK + kk]) = p;
        }
    };

    const int nk = K >> 5;
    stage(0, 0);
    __syncthreads();
    int buf = 0;
    for (int it = 1; it < nk; ++it) {
        stage(buf ^ 1, it << 5);                     // prefetch next tile
        wmma_step(As + buf * ASZ, Bs + buf * BSZ,    // compute current tile
                  wm, wn, r, hh, a00, a01, a10, a11);
        __syncthreads();
        buf ^= 1;
    }
    wmma_step(As + buf * ASZ, Bs + buf * BSZ, wm, wn, r, hh, a00, a01, a10, a11);

    // store: VGPR i -> M = i + 8*hh, N = r
    #pragma unroll
    for (int i = 0; i < 8; ++i) {
        const long row0 = (long)m0 + wm + i + 8 * hh;
        const long row1 = row0 + 16;
        float v00 = a00[i], v01 = a01[i], v10 = a10[i], v11 = a11[i];
        if (act == 1) {  // exact GELU
            v00 = 0.5f * v00 * (1.0f + erff(v00 * 0.70710678f));
            v01 = 0.5f * v01 * (1.0f + erff(v01 * 0.70710678f));
            v10 = 0.5f * v10 * (1.0f + erff(v10 * 0.70710678f));
            v11 = 0.5f * v11 * (1.0f + erff(v11 * 0.70710678f));
        }
        C[row0 * (size_t)N + n0 + wn + r]      = v00;
        C[row0 * (size_t)N + n0 + wn + 16 + r] = v01;
        C[row1 * (size_t)N + n0 + wn + r]      = v10;
        C[row1 * (size_t)N + n0 + wn + 16 + r] = v11;
    }
}

// ======================= Embedding =========================================
__device__ __forceinline__ float sintab(int pos, int d) {
    float i = (float)(d & ~1);
    float ang = (float)pos * expf(i * (-9.210340371976184f / 512.0f)); // /1e4^(i/512)
    return (d & 1) ? cosf(ang) : sinf(ang);
}

__global__ void k_embed(const float* __restrict__ x, const int* __restrict__ mark,
                        const float* __restrict__ w, float* __restrict__ out,
                        int L, int Cin)
{
    const int token = blockIdx.x;             // b*L + l
    const int b = token / L, l = token % L;
    const int lm1 = (l - 1 + L) % L, lp1 = (l + 1) % L;
    const int m0 = mark[token * 4 + 0], m1 = mark[token * 4 + 1];
    const int m2 = mark[token * 4 + 2], m3 = mark[token * 4 + 3];
    const int li[3] = {lm1, l, lp1};
    for (int d = threadIdx.x; d < 512; d += blockDim.x) {
        float acc = 0.f;
        for (int di = 0; di < 3; ++di) {
            const float* xr = x + ((size_t)b * L + li[di]) * Cin;
            const float* wr = w + (size_t)di * Cin * 512 + d;
            for (int c = 0; c < Cin; ++c) acc += xr[c] * wr[c * 512];
        }
        acc += sintab(l, d);
        acc += sintab(m0, d) + sintab(m1, d) + sintab(m2, d) + sintab(m3, d);
        out[(size_t)token * 512 + d] = acc;
    }
}

__global__ void k_map_cnn(const float* __restrict__ m, const float* __restrict__ cw,
                          const float* __restrict__ lin, float* __restrict__ out)
{
    const int token = blockIdx.x;
    const int tid = threadIdx.x;
    __shared__ float sacc[8];
    if (tid < 8) sacc[tid] = 0.f;
    __syncthreads();
    const float* mp = m + (size_t)token * 576;      // 24*24
    for (int p = tid; p < 144; p += blockDim.x) {
        const int oy = p / 12, ox = p % 12;
        float cv[8] = {0, 0, 0, 0, 0, 0, 0, 0};
        for (int ky = 0; ky < 3; ++ky) {
            const int iy = 2 * oy + ky;
            if (iy >= 24) continue;
            for (int kx = 0; kx < 3; ++kx) {
                const int ix = 2 * ox + kx;
                if (ix >= 24) continue;
                const float xv = mp[iy * 24 + ix];
                const float* wv = cw + (ky * 3 + kx) * 8;
                #pragma unroll
                for (int c = 0; c < 8; ++c) cv[c] += xv * wv[c];
            }
        }
        #pragma unroll
        for (int c = 0; c < 8; ++c) atomicAdd(&sacc[c], fmaxf(cv[c], 0.f));
    }
    __syncthreads();
    float g[8];
    #pragma unroll
    for (int c = 0; c < 8; ++c) g[c] = sacc[c] * (1.0f / 144.0f);
    for (int d = tid; d < 512; d += blockDim.x) {
        float a = 0.f;
        #pragma unroll
        for (int c = 0; c < 8; ++c) a += g[c] * lin[c * 512 + d];
        out[(size_t)token * 512 + d] += a;
    }
}

// ======================= LayerNorm (+residual) =============================
__global__ void k_add_ln(const float* __restrict__ res, const float* __restrict__ x,
                         float* __restrict__ out)
{
    const int row = blockIdx.x, tid = threadIdx.x;
    __shared__ float r1[256], r2[256];
    const size_t base = (size_t)row * 512;
    float a = x[base + tid]       + (res ? res[base + tid]       : 0.f);
    float b = x[base + tid + 256] + (res ? res[base + tid + 256] : 0.f);
    r1[tid] = a + b;
    r2[tid] = a * a + b * b;
    __syncthreads();
    for (int st = 128; st > 0; st >>= 1) {
        if (tid < st) { r1[tid] += r1[tid + st]; r2[tid] += r2[tid + st]; }
        __syncthreads();
    }
    const float mean = r1[0] * (1.0f / 512.0f);
    const float var  = r2[0] * (1.0f / 512.0f) - mean * mean;
    const float inv  = rsqrtf(var + 1e-5f);
    out[base + tid]       = (a - mean) * inv;
    out[base + tid + 256] = (b - mean) * inv;
}

// ======================= ProbSparse attention ==============================
__device__ __forceinline__ int samp_idx(int l, int u, int Lk) {
    unsigned r = (unsigned)(l * 131) + (unsigned)u * 2654435761u + 12345u;
    r ^= r >> 13; r *= 2246822519u; r ^= r >> 16;
    return (int)(r % (unsigned)Lk);
}

__global__ void k_sparsity(const float* __restrict__ q, const float* __restrict__ k,
                           float* __restrict__ spar, int Lq, int Lk, int upart)
{
    const int idx = blockIdx.x;             // bh*Lq + l
    const int l = idx % Lq, bh = idx / Lq;
    const int h = bh & 7, b = bh >> 3;
    const float* qv = q + ((size_t)(b * Lq + l)) * 512 + h * 64;
    float mx = -1e30f, sm = 0.f;
    for (int u = threadIdx.x; u < upart; u += 32) {
        const int ki = samp_idx(l, u, Lk);
        const float* kv = k + ((size_t)(b * Lk + ki)) * 512 + h * 64;
        float dot = 0.f;
        #pragma unroll
        for (int d = 0; d < 64; ++d) dot += qv[d] * kv[d];
        mx = fmaxf(mx, dot);
        sm += dot;
    }
    #pragma unroll
    for (int o = 16; o > 0; o >>= 1) {
        mx = fmaxf(mx, __shfl_xor(mx, o));
        sm += __shfl_xor(sm, o);
    }
    if (threadIdx.x == 0) spar[idx] = mx - sm / (float)upart;
}

__global__ void k_topk(float* __restrict__ spar, int* __restrict__ top, int Lq, int nt)
{
    if (threadIdx.x != 0) return;
    const int bh = blockIdx.x;
    float* s = spar + (size_t)bh * Lq;
    for (int t = 0; t < nt; ++t) {
        float best = -1e30f; int bi = 0;
        for (int i = 0; i < Lq; ++i)
            if (s[i] > best) { best = s[i]; bi = i; }
        top[bh * 64 + t] = bi;
        s[bi] = -3e30f;
    }
}

__global__ void k_ctx_mean(const float* __restrict__ v, float* __restrict__ ctx,
                           int Lq, int Lk)
{
    const int bh = blockIdx.x, h = bh & 7, b = bh >> 3, d = threadIdx.x;
    float s = 0.f;
    for (int kk = 0; kk < Lk; ++kk) s += v[((size_t)(b * Lk + kk)) * 512 + h * 64 + d];
    s /= (float)Lk;
    for (int l = 0; l < Lq; ++l)
        ctx[(((size_t)bh * Lq) + l) * 64 + d] = s;
}

__global__ void k_ctx_cumsum(const float* __restrict__ v, float* __restrict__ ctx, int L)
{
    const int bh = blockIdx.x, h = bh & 7, b = bh >> 3, d = threadIdx.x;
    float s = 0.f;
    for (int l = 0; l < L; ++l) {
        s += v[((size_t)(b * L + l)) * 512 + h * 64 + d];
        ctx[(((size_t)bh * L) + l) * 64 + d] = s;
    }
}

__global__ void k_upd(const float* __restrict__ q, const float* __restrict__ k,
                      const float* __restrict__ v, const int* __restrict__ top,
                      float* __restrict__ ctx, int Lq, int Lk, int nt, int mask)
{
    __shared__ float sc[1024];
    __shared__ float red[64];
    const int u = blockIdx.x % nt, bh = blockIdx.x / nt;
    const int h = bh & 7, b = bh >> 3, tid = threadIdx.x;
    const int tpos = top[bh * 64 + u];
    const float* qv = q + ((size_t)(b * Lq + tpos)) * 512 + h * 64;
    for (int kk = tid; kk < Lk; kk += 64) {
        const float* kv = k + ((size_t)(b * Lk + kk)) * 512 + h * 64;
        float dot = 0.f;
        #pragma unroll
        for (int d = 0; d < 64; ++d) dot += qv[d] * kv[d];
        dot *= 0.125f;
        if (mask && kk > tpos) dot = -1e9f;
        sc[kk] = dot;
    }
    __syncthreads();
    float m = -1e30f;
    for (int kk = tid; kk < Lk; kk += 64) m = fmaxf(m, sc[kk]);
    red[tid] = m; __syncthreads();
    if (tid == 0) { float mm = red[0]; for (int i = 1; i < 64; ++i) mm = fmaxf(mm, red[i]); red[0] = mm; }
    __syncthreads();
    const float mx = red[0];
    __syncthreads();
    float s = 0.f;
    for (int kk = tid; kk < Lk; kk += 64) s += expf(sc[kk] - mx);
    red[tid] = s; __syncthreads();
    if (tid == 0) { float ss = 0.f; for (int i = 0; i < 64; ++i) ss += red[i]; red[0] = ss; }
    __syncthreads();
    const float inv = 1.0f / red[0];
    float acc = 0.f;
    for (int kk = 0; kk < Lk; ++kk)
        acc += expf(sc[kk] - mx) * v[((size_t)(b * Lk + kk)) * 512 + h * 64 + tid];
    ctx[(((size_t)bh * Lq) + tpos) * 64 + tid] = acc * inv;
}

__global__ void k_ctx_tr(const float* __restrict__ ctx, float* __restrict__ att,
                         int L, int total)
{
    const int idx = blockIdx.x * blockDim.x + threadIdx.x;
    if (idx >= total) return;
    const int d = idx & 63;
    const int l = (idx >> 6) % L;
    const int h = (idx / (64 * L)) & 7;
    const int b = idx / (64 * L * 8);
    att[((size_t)(b * L + l)) * 512 + h * 64 + d] = ctx[idx];
}

__global__ void k_fullattn(const float* __restrict__ q, const float* __restrict__ k,
                           const float* __restrict__ v, float* __restrict__ att,
                           int Lq, int Lk)
{
    __shared__ float sc[1024];
    __shared__ float red[64];
    const int lq = blockIdx.x % Lq, bh = blockIdx.x / Lq;
    const int h = bh & 7, b = bh >> 3, tid = threadIdx.x;
    const float* qv = q + ((size_t)(b * Lq + lq)) * 512 + h * 64;
    for (int kk = tid; kk < Lk; kk += 64) {
        const float* kv = k + ((size_t)(b * Lk + kk)) * 512 + h * 64;
        float dot = 0.f;
        #pragma unroll
        for (int d = 0; d < 64; ++d) dot += qv[d] * kv[d];
        sc[kk] = dot * 0.125f;
    }
    __syncthreads();
    float m = -1e30f;
    for (int kk = tid; kk < Lk; kk += 64) m = fmaxf(m, sc[kk]);
    red[tid] = m; __syncthreads();
    if (tid == 0) { float mm = red[0]; for (int i = 1; i < 64; ++i) mm = fmaxf(mm, red[i]); red[0] = mm; }
    __syncthreads();
    const float mx = red[0];
    __syncthreads();
    float s = 0.f;
    for (int kk = tid; kk < Lk; kk += 64) s += expf(sc[kk] - mx);
    red[tid] = s; __syncthreads();
    if (tid == 0) { float ss = 0.f; for (int i = 0; i < 64; ++i) ss += red[i]; red[0] = ss; }
    __syncthreads();
    const float inv = 1.0f / red[0];
    float acc = 0.f;
    for (int kk = 0; kk < Lk; ++kk)
        acc += expf(sc[kk] - mx) * v[((size_t)(b * Lk + kk)) * 512 + h * 64 + tid];
    att[((size_t)(b * Lq + lq)) * 512 + h * 64 + tid] = acc * inv;
}

// ======================= Distil helpers ====================================
__global__ void k_pad(const float* __restrict__ x, float* __restrict__ xp,
                      int L, int total)
{
    const int idx = blockIdx.x * blockDim.x + threadIdx.x;
    if (idx >= total) return;
    const int c = idx & 511;
    const int li = (idx >> 9) % (L + 2);
    const int b = idx / (512 * (L + 2));
    const int src = (li == 0) ? (L - 1) : (li == L + 1 ? 0 : li - 1);
    xp[idx] = x[((size_t)(b * L + src)) * 512 + c];
}

__global__ void k_bn(const float* __restrict__ y, float* __restrict__ mean,
                     float* __restrict__ var, int M)
{
    const int c = blockIdx.x, tid = threadIdx.x;
    __shared__ float r1[256], r2[256];
    float s = 0.f, s2 = 0.f;
    for (int m = tid; m < M; m += 256) {
        const float v = y[(size_t)m * 512 + c];
        s += v; s2 += v * v;
    }
    r1[tid] = s; r2[tid] = s2;
    __syncthreads();
    for (int st = 128; st > 0; st >>= 1) {
        if (tid < st) { r1[tid] += r1[tid + st]; r2[tid] += r2[tid + st]; }
        __syncthreads();
    }
    if (tid == 0) {
        const float mu = r1[0] / (float)M;
        mean[c] = mu;
        var[c]  = r2[0] / (float)M - mu * mu;
    }
}

__global__ void k_elupool(const float* __restrict__ y, const float* __restrict__ mean,
                          const float* __restrict__ var, float* __restrict__ out,
                          int L, int Lo, int total)
{
    const int idx = blockIdx.x * blockDim.x + threadIdx.x;
    if (idx >= total) return;
    const int c = idx & 511;
    const int lo = (idx >> 9) % Lo;
    const int b = idx / (512 * Lo);
    const float mu = mean[c];
    const float inv = rsqrtf(var[c] + 1e-5f);
    float mx = -1e30f;
    for (int t = -1; t <= 1; ++t) {
        const int li = 2 * lo + t;
        if (li < 0 || li >= L) continue;
        float v = (y[((size_t)(b * L + li)) * 512 + c] - mu) * inv;
        v = (v > 0.f) ? v : (expf(v) - 1.f);
        mx = fmaxf(mx, v);
    }
    out[((size_t)(b * Lo + lo)) * 512 + c] = mx;
}

// ======================= Projection ========================================
__global__ void k_proj(const float* __restrict__ d, const float* __restrict__ pw,
                       const float* __restrict__ pb, float* __restrict__ out)
{
    const int idx = blockIdx.x * blockDim.x + threadIdx.x;
    if (idx >= 8 * 256 * 4) return;
    const int c = idx & 3;
    const int t = (idx >> 2) & 255;
    const int b = idx >> 10;
    const size_t row = (size_t)b * 512 + 256 + t;   // last OUT_LEN of DEC_LEN=512
    float s = pb[c];
    for (int k = 0; k < 512; ++k) s += d[row * 512 + k] * pw[k * 4 + c];
    out[idx] = s;
}

// ======================= Host orchestration ================================
static inline int topn_for(int L) {
    int v = (int)(5.0 * ceil(log((double)L)));
    return v < L ? v : L;
}

extern "C" void kernel_launch(void* const* d_in, const int* in_sizes, int n_in,
                              void* d_out, int out_size, void* d_ws, size_t ws_size,
                              hipStream_t stream)
{
    (void)in_sizes; (void)n_in; (void)out_size; (void)ws_size;
    const float* x_enc        = (const float*)d_in[0];
    const float* x_map        = (const float*)d_in[1];
    const float* y_map        = (const float*)d_in[2];
    const float* x_dec        = (const float*)d_in[3];
    const float* enc_tok_w    = (const float*)d_in[4];
    const float* dec_tok_w    = (const float*)d_in[5];
    const float* enc_map_conv = (const float*)d_in[6];
    const float* enc_map_lin  = (const float*)d_in[7];
    const float* dec_map_conv = (const float*)d_in[8];
    const float* dec_map_lin  = (const float*)d_in[9];
    const float* enc_attn_w   = (const float*)d_in[10];
    const float* enc_ffn1     = (const float*)d_in[11];
    const float* enc_ffn2     = (const float*)d_in[12];
    const float* dist_w       = (const float*)d_in[13];
    const float* dec_self_w   = (const float*)d_in[14];
    const float* dec_cross_w  = (const float*)d_in[15];
    const float* dec_ffn1     = (const float*)d_in[16];
    const float* dec_ffn2     = (const float*)d_in[17];
    const float* proj_w       = (const float*)d_in[18];
    const float* proj_b       = (const float*)d_in[19];
    const int*   mark_enc     = (const int*)d_in[20];
    const int*   mark_dec     = (const int*)d_in[21];

    float* ws = (float*)d_ws;
    const size_t S = (size_t)8 * 1026 * 512;     // slot: fits [8,1026,512]
    float* H    = ws + 0 * S;   // encoder hidden (and final enc output)
    float* Q    = ws + 1 * S;
    float* Kb   = ws + 2 * S;
    float* Vb   = ws + 3 * S;
    float* CTX  = ws + 4 * S;   // [B,H,L,64]
    float* ATT  = ws + 5 * S;   // transposed attn out; reused as circular pad buf
    float* T0   = ws + 6 * S;
    float* T1   = ws + 7 * S;
    float* DD   = ws + 8 * S;   // decoder hidden
    float* MISC = ws + 9 * S;
    float* SPAR = MISC;                           // 64*1024
    int*   TOP  = (int*)(MISC + 65536);           // 64*64
    float* BNM  = MISC + 65536 + 4096;            // 512
    float* BNV  = BNM + 512;                      // 512

    const size_t WSTRIDE = 512 * 512;

    auto gemm = [&](const float* A, const float* W, float* C, int M, int N, int K,
                    int lda, int rpb, int extra, int act) {
        dim3 g(N / 64, M / 128);
        k_gemm<<<g, 256, 0, stream>>>(A, W, C, M, N, K, lda, rpb, extra, act);
    };

    // ---------------- Encoder ----------------
    k_embed<<<8 * 1024, 256, 0, stream>>>(x_enc, mark_enc, enc_tok_w, H, 1024, 4);
    k_map_cnn<<<8 * 1024, 256, 0, stream>>>(x_map, enc_map_conv, enc_map_lin, H);

    int L = 1024;
    for (int l = 0; l < 3; ++l) {
        const int M = 8 * L;
        const float* Wl = enc_attn_w + (size_t)l * 4 * WSTRIDE;
        gemm(H, Wl + 0 * WSTRIDE, Q,  M, 512, 512, 512, M, 0, 0);
        gemm(H, Wl + 1 * WSTRIDE, Kb, M, 512, 512, 512, M, 0, 0);
        gemm(H, Wl + 2 * WSTRIDE, Vb, M, 512, 512, 512, M, 0, 0);
        const int nt = topn_for(L);
        k_sparsity<<<64 * L, 32, 0, stream>>>(Q, Kb, SPAR, L, L, nt);
        k_topk<<<64, 32, 0, stream>>>(SPAR, TOP, L, nt);
        k_ctx_mean<<<64, 64, 0, stream>>>(Vb, CTX, L, L);
        k_upd<<<64 * nt, 64, 0, stream>>>(Q, Kb, Vb, TOP, CTX, L, L, nt, 0);
        const int tot = M * 512;
        k_ctx_tr<<<(tot + 255) / 256, 256, 0, stream>>>(CTX, ATT, L, tot);
        gemm(ATT, Wl + 3 * WSTRIDE, T0, M, 512, 512, 512, M, 0, 0);
        k_add_ln<<<M, 256, 0, stream>>>(H, T0, H);
        // FFN
        gemm(H,  enc_ffn1 + (size_t)l * WSTRIDE, T0, M, 512, 512, 512, M, 0, 1);
        gemm(T0, enc_ffn2 + (size_t)l * WSTRIDE, T1, M, 512, 512, 512, M, 0, 0);
        k_add_ln<<<M, 256, 0, stream>>>(H, T1, H);
        // distil
        if (l < 2) {
            const int ptot = 8 * (L + 2) * 512;
            k_pad<<<(ptot + 255) / 256, 256, 0, stream>>>(H, ATT, L, ptot);
            // implicit im2col over padded buffer: K=1536, row remap +2/batch
            gemm(ATT, dist_w + (size_t)l * 3 * WSTRIDE, T0, M, 512, 1536, 512, L, 2, 0);
            k_bn<<<512, 256, 0, stream>>>(T0, BNM, BNV, M);
            const int Lo = L / 2;
            const int etot = 8 * Lo * 512;
            k_elupool<<<(etot + 255) / 256, 256, 0, stream>>>(T0, BNM, BNV, H, L, Lo, etot);
            L = Lo;
        }
    }
    const int Le = L;           // 256
    const int Me = 8 * Le;
    k_add_ln<<<Me, 256, 0, stream>>>(nullptr, H, H);   // final encoder LN

    // ---------------- Decoder ----------------
    const int Ld = 512, Md = 8 * 512;
    k_embed<<<8 * 512, 256, 0, stream>>>(x_dec, mark_dec, dec_tok_w, DD, 512, 4);
    k_map_cnn<<<8 * 512, 256, 0, stream>>>(y_map, dec_map_conv, dec_map_lin, DD);

    for (int l = 0; l < 2; ++l) {
        // --- masked ProbSparse self-attention (mix=True) ---
        const float* Ws = dec_self_w + (size_t)l * 4 * WSTRIDE;
        gemm(DD, Ws + 0 * WSTRIDE, Q,  Md, 512, 512, 512, Md, 0, 0);
        gemm(DD, Ws + 1 * WSTRIDE, Kb, Md, 512, 512, 512, Md, 0, 0);
        gemm(DD, Ws + 2 * WSTRIDE, Vb, Md, 512, 512, 512, Md, 0, 0);
        const int nt = topn_for(Ld);   // 35
        k_sparsity<<<64 * Ld, 32, 0, stream>>>(Q, Kb, SPAR, Ld, Ld, nt);
        k_topk<<<64, 32, 0, stream>>>(SPAR, TOP, Ld, nt);
        k_ctx_cumsum<<<64, 64, 0, stream>>>(Vb, CTX, Ld);
        k_upd<<<64 * nt, 64, 0, stream>>>(Q, Kb, Vb, TOP, CTX, Ld, Ld, nt, 1);
        // mix quirk: [B,H,L,64] reinterpreted as [B,L,512] -> feed CTX directly
        gemm(CTX, Ws + 3 * WSTRIDE, T0, Md, 512, 512, 512, Md, 0, 0);
        k_add_ln<<<Md, 256, 0, stream>>>(DD, T0, DD);

        // --- full cross-attention on encoder output ---
        const float* Wc = dec_cross_w + (size_t)l * 4 * WSTRIDE;
        gemm(DD, Wc + 0 * WSTRIDE, Q,  Md, 512, 512, 512, Md, 0, 0);
        gemm(H,  Wc + 1 * WSTRIDE, Kb, Me, 512, 512, 512, Me, 0, 0);
        gemm(H,  Wc + 2 * WSTRIDE, Vb, Me, 512, 512, 512, Me, 0, 0);
        k_fullattn<<<64 * Ld, 64, 0, stream>>>(Q, Kb, Vb, ATT, Ld, Le);
        gemm(ATT, Wc + 3 * WSTRIDE, T0, Md, 512, 512, 512, Md, 0, 0);
        k_add_ln<<<Md, 256, 0, stream>>>(DD, T0, DD);

        // --- FFN ---
        gemm(DD, dec_ffn1 + (size_t)l * WSTRIDE, T0, Md, 512, 512, 512, Md, 0, 1);
        gemm(T0, dec_ffn2 + (size_t)l * WSTRIDE, T1, Md, 512, 512, 512, Md, 0, 0);
        k_add_ln<<<Md, 256, 0, stream>>>(DD, T1, DD);
    }
    k_add_ln<<<Md, 256, 0, stream>>>(nullptr, DD, DD);  // final decoder LN

    // projection + slice last OUT_LEN
    k_proj<<<(8 * 256 * 4 + 255) / 256, 256, 0, stream>>>(DD, proj_w, proj_b,
                                                          (float*)d_out);
}
